// AttentionHead_76218489635437
// MI455X (gfx1250) — matrix operand order repaired and checked by reading, not compiled, not run on hardware
//
#include <hip/hip_runtime.h>
#include <hip/hip_bf16.h>

typedef __attribute__((ext_vector_type(16))) __bf16 bf16x16;
typedef __attribute__((ext_vector_type(8)))  float  f32x8;

union Frag {
    bf16x16 bf;
    uint4 q[2];
    unsigned int u32[8];
    unsigned short us[16];
};

// Native packed f32x2 -> bf16x2 (RNE). Verified: assembler accepts on gfx1250.
__device__ __forceinline__ unsigned int cvt_pk_bf16(float lo, float hi) {
    unsigned int r;
    asm("v_cvt_pk_bf16_f32 %0, %1, %2" : "=v"(r) : "v"(lo), "v"(hi));
    return r;
}

__device__ __forceinline__ unsigned short f2bf(float x) {
    return (unsigned short)(cvt_pk_bf16(x, x) & 0xFFFFu);
}

#define S_LEN 4096
#define D_IN  1024
#define KT_STRIDE 144            // 128B row + 16B pad (bank spread, 16B aligned)
#define PT_STRIDE 144
#define KV_BUF (2 * 64 * KT_STRIDE)   // one double-buffer slot: K tile + V tile

// Scale folded into Q: (1/sqrt(64)) * log2(e)
#define Q_SCALE 0.18033688011112042f

// ---------------------------------------------------------------------------
// Phase 0: transpose + convert W [1024][64] f32 -> Wt [64][1024] bf16 (x3)
// ---------------------------------------------------------------------------
__global__ void wt_kernel(const float* __restrict__ Wq,
                          const float* __restrict__ Wk,
                          const float* __restrict__ Wv,
                          unsigned short* __restrict__ Wt) {
    int t = blockIdx.x * blockDim.x + threadIdx.x;   // 0 .. 3*65536-1
    int w   = t >> 16;
    int rem = t & 65535;            // rem = k*64 + d
    int k = rem >> 6;
    int d = rem & 63;
    const float* W = (w == 0) ? Wq : ((w == 1) ? Wk : Wv);
    Wt[(size_t)w * 65536 + (size_t)d * D_IN + k] = f2bf(W[rem]);
}

// ---------------------------------------------------------------------------
// Phase 1: projections X[16384][1024] @ W -> bf16 Q,K (row major), Vt (transposed)
// One wave per 16 output rows. K loop unrolled x2 with *statically named*
// ping-pong B fragments so everything stays in registers (no scratch spills).
// blockDim=128 (4 waves), grid=(256, 3).
// ---------------------------------------------------------------------------
__global__ __launch_bounds__(128) void proj_kernel(
    const float* __restrict__ Xq, const float* __restrict__ Xk, const float* __restrict__ Xv,
    const unsigned short* __restrict__ Wt,
    const float* __restrict__ bq, const float* __restrict__ bk, const float* __restrict__ bv,
    unsigned short* __restrict__ Qbf, unsigned short* __restrict__ Kbf,
    unsigned short* __restrict__ Vt)
{
    int proj = blockIdx.y;
    int wave = threadIdx.x >> 5, lane = threadIdx.x & 31;
    int m = lane & 15, h = lane >> 4;
    int row0 = (blockIdx.x * 4 + wave) * 16;

    const float* X    = (proj == 0) ? Xq : ((proj == 1) ? Xk : Xv);
    const float* bias = (proj == 0) ? bq : ((proj == 1) ? bk : bv);
    const unsigned short* wt = Wt + (size_t)proj * 65536;
    float scale = (proj == 0) ? Q_SCALE : 1.0f;

    const float* arow = X + (size_t)(row0 + m) * D_IN;

    float4 f[4];              // raw A f32 staging (constant-indexed only)
    Frag bA[4], bB[4];        // explicit ping-pong B fragments

    auto load_chunkA = [&](int kk) {
#pragma unroll
        for (int g = 0; g < 2; ++g) {
            const float4* p = (const float4*)(arow + kk + h * 16 + g * 8);
            f[2 * g]     = p[0];
            f[2 * g + 1] = p[1];
        }
    };
    auto load_chunkB = [&](int kk, Frag (&bf4)[4]) {
#pragma unroll
        for (int n = 0; n < 4; ++n) {   // lane holds column n*16+m, K = h*16+0..15
            const char* bp = (const char*)(wt + (size_t)(n * 16 + m) * D_IN + kk) + h * 32;
            bf4[n].q[0] = *(const uint4*)(bp);
            bf4[n].q[1] = *(const uint4*)(bp + 16);
        }
    };
    auto conv_a = [&]() {
        Frag a;   // elems 0..15 -> K=h*16+0..15
#pragma unroll
        for (int g = 0; g < 2; ++g) {
            a.u32[4 * g + 0] = cvt_pk_bf16(f[2 * g].x,     f[2 * g].y);
            a.u32[4 * g + 1] = cvt_pk_bf16(f[2 * g].z,     f[2 * g].w);
            a.u32[4 * g + 2] = cvt_pk_bf16(f[2 * g + 1].x, f[2 * g + 1].y);
            a.u32[4 * g + 3] = cvt_pk_bf16(f[2 * g + 1].z, f[2 * g + 1].w);
        }
        return a;
    };

    f32x8 acc[4] = {};
    load_chunkA(0);
    load_chunkB(0, bA);

    const int NC = D_IN / 32;   // 32 chunks, even
    for (int ic = 0; ic < NC; ic += 2) {
        // ---- even chunk: compute with bA, prefetch (ic+1) into bB ----
        {
            Frag a = conv_a();                 // chunk ic (arrived last iter)
            load_chunkA((ic + 1) * 32);
            load_chunkB((ic + 1) * 32, bB);    // in flight across WMMAs below
#pragma unroll
            for (int n = 0; n < 4; ++n)
                acc[n] = __builtin_amdgcn_wmma_f32_16x16x32_bf16(
                    false, a.bf, false, bA[n].bf, (short)0, acc[n], false, false);
        }
        // ---- odd chunk: compute with bB, prefetch (ic+2) into bA ----
        {
            Frag a = conv_a();                 // chunk ic+1
            if (ic + 2 < NC) {
                load_chunkA((ic + 2) * 32);
                load_chunkB((ic + 2) * 32, bA);
            }
#pragma unroll
            for (int n = 0; n < 4; ++n)
                acc[n] = __builtin_amdgcn_wmma_f32_16x16x32_bf16(
                    false, a.bf, false, bB[n].bf, (short)0, acc[n], false, false);
        }
    }

    // bias + scale + store. C layout: row M = r + 8*h, col N = n*16 + m.
#pragma unroll
    for (int n = 0; n < 4; ++n) {
        int dim = n * 16 + m;
        float bb = bias[dim];
#pragma unroll
        for (int r = 0; r < 8; ++r) {
            int row = row0 + r + 8 * h;
            unsigned short bits = f2bf((acc[n][r] + bb) * scale);
            if (proj == 0)      Qbf[(size_t)row * 64 + dim] = bits;
            else if (proj == 1) Kbf[(size_t)row * 64 + dim] = bits;
            else {
                int b_ = row >> 12, s_ = row & (S_LEN - 1);
                Vt[((size_t)b_ * 64 + dim) * S_LEN + s_] = bits;
            }
        }
    }
}

// ---------------------------------------------------------------------------
// Phase 2: flash attention, double-buffered K/V chunks in LDS.
// blockDim=128 (4 waves = 64 q rows), grid=(64, 4).
// ---------------------------------------------------------------------------
__global__ __launch_bounds__(128) void attn_kernel(
    const unsigned short* __restrict__ Qbf,
    const unsigned short* __restrict__ Kbf,
    const unsigned short* __restrict__ Vt,
    float* __restrict__ out)
{
    __shared__ __align__(16) char smem[2 * KV_BUF + 4 * 16 * PT_STRIDE];

    int wave = threadIdx.x >> 5, lane = threadIdx.x & 31;
    int m = lane & 15, h = lane >> 4;
    char* pT = smem + 2 * KV_BUF + wave * 16 * PT_STRIDE;  // wave-private P

    int b  = blockIdx.y;
    int q0 = blockIdx.x * 64 + wave * 16;

    // Q resident in registers: two K=32 A-fragments (row = q0+m)
    Frag aq[2];
    {
        const char* qb = (const char*)(Qbf + ((size_t)b * S_LEN + q0 + m) * 64) + h * 32;
        aq[0].q[0] = *(const uint4*)(qb);
        aq[0].q[1] = *(const uint4*)(qb + 16);
        aq[1].q[0] = *(const uint4*)(qb + 64);
        aq[1].q[1] = *(const uint4*)(qb + 80);
    }

    f32x8 O[4] = {};
    float mrow[8], lsum[8];
#pragma unroll
    for (int r = 0; r < 8; ++r) { mrow[r] = -1e30f; lsum[r] = 0.0f; }

    int t   = threadIdx.x;
    int ldr = t >> 1;            // 0..63: key row (K tile) / dim row (Vt tile)
    int ldo = (t & 1) * 64;      // 64B halves of the 128B row
    const char* gkbase = (const char*)(Kbf + ((size_t)b * S_LEN + ldr) * 64) + ldo;
    const char* gvbase = (const char*)(Vt + ((size_t)b * 64 + ldr) * S_LEN) + ldo;

    uint4 rk[4], rv[4];          // register staging for next chunk
    auto stage_load = [&](int kb) {
        const uint4* gk = (const uint4*)(gkbase + (size_t)kb * 128);
        const uint4* gv = (const uint4*)(gvbase + (size_t)kb * 2);
#pragma unroll
        for (int i = 0; i < 4; ++i) { rk[i] = gk[i]; rv[i] = gv[i]; }
    };
    auto stage_store = [&](int buf) {
        char* lk = smem + buf * KV_BUF + ldr * KT_STRIDE + ldo;
        char* lv = lk + 64 * KT_STRIDE;
#pragma unroll
        for (int i = 0; i < 4; ++i) {
            ((uint4*)lk)[i] = rk[i];
            ((uint4*)lv)[i] = rv[i];
        }
    };

    stage_load(0);
    stage_store(0);
    __syncthreads();

    const int NCH = S_LEN / 64;
    for (int ci = 0; ci < NCH; ++ci) {
        if (ci + 1 < NCH) stage_load((ci + 1) * 64);   // overlap with compute

        char* kT = smem + (ci & 1) * KV_BUF;
        char* vT = kT + 64 * KT_STRIDE;

        // Scores: 16 q-rows x 64 keys, K-dim = 64 (2 WMMAs per 16-key tile)
        f32x8 c4[4];
#pragma unroll
        for (int j = 0; j < 4; ++j) {
            Frag b0, b1;   // B layout: lane holds key column j*16+m, d = h*16+0..15
            const char* kp = kT + (j * 16 + m) * KT_STRIDE + h * 32;
            b0.q[0] = *(const uint4*)(kp);
            b0.q[1] = *(const uint4*)(kp + 16);
            b1.q[0] = *(const uint4*)(kp + 64);
            b1.q[1] = *(const uint4*)(kp + 80);
            f32x8 c = {};
            c = __builtin_amdgcn_wmma_f32_16x16x32_bf16(false, aq[0].bf, false, b0.bf, (short)0, c, false, false);
            c = __builtin_amdgcn_wmma_f32_16x16x32_bf16(false, aq[1].bf, false, b1.bf, (short)0, c, false, false);
            c4[j] = c;
        }

        // Online softmax (base-2; log2e folded into Q). Row M=r+8h lives in one
        // 16-lane half -> shfl_xor masks 1,2,4,8 reduce without crossing halves.
        float alpha[8];
#pragma unroll
        for (int r = 0; r < 8; ++r) {
            float t0 = fmaxf(fmaxf(c4[0][r], c4[1][r]), fmaxf(c4[2][r], c4[3][r]));
#pragma unroll
            for (int s = 1; s < 16; s <<= 1) t0 = fmaxf(t0, __shfl_xor(t0, s, 32));
            float mn = fmaxf(mrow[r], t0);
            alpha[r] = exp2f(mrow[r] - mn);
            mrow[r] = mn;
        }
#pragma unroll
        for (int r = 0; r < 8; ++r) {
            float rs = 0.0f;
#pragma unroll
            for (int j = 0; j < 4; ++j) {
                float p = exp2f(c4[j][r] - mrow[r]);
                c4[j][r] = p;
                rs += p;
            }
#pragma unroll
            for (int s = 1; s < 16; s <<= 1) rs += __shfl_xor(rs, s, 32);
            lsum[r] = lsum[r] * alpha[r] + rs;
        }
#pragma unroll
        for (int n = 0; n < 4; ++n)
#pragma unroll
            for (int r = 0; r < 8; ++r) O[n][r] *= alpha[r];

        // P (C layout) -> LDS row-major bf16 [16 rows][64 keys]
#pragma unroll
        for (int j = 0; j < 4; ++j)
#pragma unroll
            for (int r = 0; r < 8; ++r)
                *(unsigned short*)(pT + (r + 8 * h) * PT_STRIDE + (j * 16 + m) * 2) =
                    f2bf(c4[j][r]);

        asm volatile("s_wait_dscnt 0x0" ::: "memory");  // same-wave LDS RAW

        // O += P @ V  (A = P reloaded in A layout, B = Vt rows, 2 K-chunks)
#pragma unroll
        for (int kc = 0; kc < 2; ++kc) {
            Frag pa;
            const char* pp = pT + m * PT_STRIDE + kc * 64 + h * 32;
            pa.q[0] = *(const uint4*)(pp);
            pa.q[1] = *(const uint4*)(pp + 16);
#pragma unroll
            for (int n = 0; n < 4; ++n) {
                Frag vb;
                const char* vp = vT + (n * 16 + m) * KT_STRIDE + kc * 64 + h * 32;
                vb.q[0] = *(const uint4*)(vp);
                vb.q[1] = *(const uint4*)(vp + 16);
                O[n] = __builtin_amdgcn_wmma_f32_16x16x32_bf16(
                    false, pa.bf, false, vb.bf, (short)0, O[n], false, false);
            }
        }

        // Writers target the *other* buffer; one barrier per iteration suffices.
        if (ci + 1 < NCH) stage_store((ci + 1) & 1);
        __syncthreads();
    }

    // Normalize and store f32 output [B,S,64]
#pragma unroll
    for (int r = 0; r < 8; ++r) {
        float inv = 1.0f / lsum[r];
        int row = q0 + r + 8 * h;
#pragma unroll
        for (int n = 0; n < 4; ++n)
            out[((size_t)b * S_LEN + row) * 64 + n * 16 + m] = O[n][r] * inv;
    }
}

// ---------------------------------------------------------------------------
extern "C" void kernel_launch(void* const* d_in, const int* in_sizes, int n_in,
                              void* d_out, int out_size, void* d_ws, size_t ws_size,
                              hipStream_t stream) {
    const float* query = (const float*)d_in[0];
    const float* key_  = (const float*)d_in[1];
    const float* value = (const float*)d_in[2];
    const float* Wq = (const float*)d_in[3];
    const float* bq = (const float*)d_in[4];
    const float* Wk = (const float*)d_in[5];
    const float* bk = (const float*)d_in[6];
    const float* Wv = (const float*)d_in[7];
    const float* bv = (const float*)d_in[8];

    char* ws = (char*)d_ws;
    unsigned short* Wt  = (unsigned short*)(ws);                         // 384 KB
    unsigned short* Qbf = (unsigned short*)(ws + (1 << 19));             // 2 MB
    unsigned short* Kbf = (unsigned short*)(ws + (1 << 19) + (1 << 21)); // 2 MB
    unsigned short* Vt  = (unsigned short*)(ws + (1 << 19) + (2 << 21)); // 2 MB

    wt_kernel<<<768, 256, 0, stream>>>(Wq, Wk, Wv, Wt);
    proj_kernel<<<dim3(256, 3), 128, 0, stream>>>(query, key_, value, Wt,
                                                  bq, bk, bv, Qbf, Kbf, Vt);
    attn_kernel<<<dim3(64, 4), 128, 0, stream>>>(Qbf, Kbf, Vt, (float*)d_out);
}
